// DualDomainCrossAttention_17952963297505
// MI455X (gfx1250) — matrix operand order repaired and checked by reading, not compiled, not run on hardware
//
#include <hip/hip_runtime.h>

#define DEV __device__ __forceinline__

// ---- problem constants ----
#define BB   2
#define CC   256
#define HH   56
#define WWW  56
#define NN   3136      // H*W
#define MM   6272      // B*N
#define NHH  8
#define HDD  32
#define SCALE_F 0.17677669529663687f   // 1/sqrt(32)

// ---- CDNA5 WMMA types (trivial types only, safe inside unions) ----
typedef __attribute__((ext_vector_type(16))) __bf16        v16bf;
typedef __attribute__((ext_vector_type(8)))  float         v8f;
typedef __attribute__((ext_vector_type(4)))  unsigned int  v4u;

union FragBF {
    v16bf          v;
    unsigned short s[16];
    v4u            u[2];
};

DEV unsigned short f2bf(float x) {
    unsigned int u = __float_as_uint(x);
    u += 0x7fffu + ((u >> 16) & 1u);           // round-to-nearest-even
    return (unsigned short)(u >> 16);
}

DEV v8f vzero8() {
    v8f z;
#pragma unroll
    for (int r = 0; r < 8; ++r) z[r] = 0.0f;
    return z;
}

// LDS byte offset of a shared-memory pointer (flat addr low 32 bits = LDS_ADDR)
DEV unsigned int lds_off(const void* p) {
    return (unsigned int)(unsigned long long)(uintptr_t)p;
}

// ---------------------------------------------------------------------------
// f32 -> bf16 flat convert (weights, activations)
// ---------------------------------------------------------------------------
__global__ void k_f32_to_bf16(const float* __restrict__ in,
                              unsigned short* __restrict__ out, int n) {
    int i = blockIdx.x * blockDim.x + threadIdx.x;
    if (i < n) out[i] = f2bf(in[i]);
}

// ---------------------------------------------------------------------------
// [B,C,H,W] f32  ->  [B*N, C] bf16 (channels-last for GEMM A operands)
// ---------------------------------------------------------------------------
__global__ void k_nchw_to_nc(const float* __restrict__ in,
                             unsigned short* __restrict__ out) {
    int i = blockIdx.x * blockDim.x + threadIdx.x;
    if (i >= BB * CC * NN) return;
    int s = i % NN;
    int c = (i / NN) % CC;
    int b = i / (NN * CC);
    out[((size_t)(b * NN + s)) * CC + c] = f2bf(in[i]);
}

// ---------------------------------------------------------------------------
// DFT-56 twiddle table:  F[a][b] = exp(-2*pi*i*a*b/56) / sqrt(56)
// ---------------------------------------------------------------------------
__global__ void k_build_dft(float* __restrict__ Fr, float* __restrict__ Fi) {
    int i = blockIdx.x * blockDim.x + threadIdx.x;
    if (i >= HH * HH) return;
    int a = i / HH, b = i % HH;
    float ang = -2.0f * 3.14159265358979323846f * (float)((a * b) % HH) / (float)HH;
    float s, c;
    __sincosf(ang, &s, &c);
    const float inv = 0.13363062095621219f;    // 1/sqrt(56)
    Fr[i] = c * inv;
    Fi[i] = s * inv;
}

// ---------------------------------------------------------------------------
// Generic bf16 WMMA GEMM:  C[M,O] (f32) = A[M,K](bf16) * W[O,K](bf16)^T (+bias)
// Block = 256 threads = 8 waves laid out 4(M) x 2(O); block tile 64x32.
// Each wave: one 16x16 D tile via v_wmma_f32_16x16x32_bf16 over K.
// ---------------------------------------------------------------------------
__global__ void k_gemm_bf16(const unsigned short* __restrict__ A, int lda,
                            const unsigned short* __restrict__ W, int ldw,
                            float* __restrict__ Cmat, int ldc,
                            const float* __restrict__ bias,
                            int K, int accum) {
    const int lane = threadIdx.x & 31;
    const int wid  = threadIdx.x >> 5;
    const int wm   = wid & 3;
    const int wo   = wid >> 2;
    const int m0   = blockIdx.x * 64 + wm * 16;
    const int o0   = blockIdx.y * 32 + wo * 16;
    const int half = lane >> 4;
    const int l15  = lane & 15;

    const unsigned short* Arow = A + (size_t)(m0 + l15) * lda;
    const unsigned short* Wrow = W + (size_t)(o0 + l15) * ldw;

    v8f acc;
    if (accum) {
#pragma unroll
        for (int r = 0; r < 8; ++r)
            acc[r] = Cmat[(size_t)(m0 + r + 8 * half) * ldc + o0 + l15];
    } else {
        acc = vzero8();
    }

    for (int k = 0; k < K; k += 32) {
        __builtin_prefetch(Arow + k + 128, 0, 0);
        FragBF fa, fb;
        // A 16x32: lanes0-15 K0-7 & K16-23 ; lanes16-31 K8-15 & K24-31
        fa.u[0] = *(const v4u*)(Arow + k + half * 8);
        fa.u[1] = *(const v4u*)(Arow + k + 16 + half * 8);
        // B 32x16: lane = column o, lanes0-15 K0-7 & K16-23 ; lanes16-31 K8-15 & K24-31
        fb.u[0] = *(const v4u*)(Wrow + k + half * 8);
        fb.u[1] = *(const v4u*)(Wrow + k + 16 + half * 8);
        acc = __builtin_amdgcn_wmma_f32_16x16x32_bf16(false, fa.v, false, fb.v,
                                                      (short)0, acc, false, false);
    }

    float bv = bias ? bias[o0 + l15] : 0.0f;
#pragma unroll
    for (int r = 0; r < 8; ++r)
        Cmat[(size_t)(m0 + r + 8 * half) * ldc + o0 + l15] = acc[r] + bv;
}

// ---------------------------------------------------------------------------
// Fused flash-attention (token branch). One wave per (b, head, 16-row q tile).
// Q: [M,256] f32, KV: [M,512] f32 (k = cols 0..255, v = cols 256..511).
// V tiles are staged HBM->LDS with CDNA5 async-tensor DMA
// (global_load_async_to_lds_b128, ASYNCcnt) so the copy overlaps the QK^T
// WMMAs + softmax VALU work, then fragments are built from LDS.
// Out: x_attn [M,256] f32.
// ---------------------------------------------------------------------------
__global__ void k_attn(const float* __restrict__ Q,
                       const float* __restrict__ KV,
                       float* __restrict__ Oout) {
    const int lane = threadIdx.x & 31;
    const int wid  = threadIdx.x >> 5;
    const int g    = blockIdx.x * 8 + wid;          // 0..3135
    const int qtile = g % 196;
    const int h     = (g / 196) % NHH;
    const int b     = g / (196 * NHH);
    const int half  = lane >> 4;
    const int l15   = lane & 15;

    __shared__ __align__(16) unsigned short pbuf[8][16 * 32];
    __shared__ __align__(16) float          vstage[8][32 * 32]; // 32 keys x 32 dims
    unsigned short* P  = pbuf[wid];
    float*          VW = vstage[wid];
    const unsigned int vw_lds = lds_off(VW) + (unsigned int)lane * 128u; // row "lane"

    // Q fragment (A, 16x32) with softmax scale folded in.
    FragBF fq;
    {
        const float* qr = Q + (size_t)(b * NN + qtile * 16 + l15) * CC + h * HDD;
#pragma unroll
        for (int j = 0; j < 8; ++j) fq.s[j]     = f2bf(qr[half * 8 + j] * SCALE_F);
#pragma unroll
        for (int j = 0; j < 8; ++j) fq.s[8 + j] = f2bf(qr[16 + half * 8 + j] * SCALE_F);
    }

    v8f o0 = vzero8(), o1 = vzero8();
    float m_run[8], l_run[8];
#pragma unroll
    for (int r = 0; r < 8; ++r) { m_run[r] = -1e30f; l_run[r] = 0.0f; }

    for (int kb = 0; kb < NN; kb += 32) {
        // ---- kick off async DMA of the 32x32 V tile into LDS (per-wave) ----
        {
            unsigned long long g64 = (unsigned long long)(uintptr_t)
                (KV + (size_t)(b * NN + kb + lane) * 512 + 256 + h * HDD);
#pragma unroll
            for (int i = 0; i < 8; ++i) {
                asm volatile("global_load_async_to_lds_b128 %0, %1, off"
                             :: "v"(vw_lds + (unsigned int)(i * 16)),
                                "v"(g64 + (unsigned long long)(i * 16))
                             : "memory");
            }
        }

        // ---- K fragments as B matrices (two 16-key sub-tiles) ----
        FragBF fk0, fk1;
        {
            const float* k0 = KV + (size_t)(b * NN + kb + l15) * 512 + h * HDD;
            const float* k1 = KV + (size_t)(b * NN + kb + 16 + l15) * 512 + h * HDD;
#pragma unroll
            for (int j = 0; j < 16; ++j) {
                fk0.s[j] = f2bf(k0[(j < 8 ? half * 8 + j : 8 + half * 8 + j)]);
                fk1.s[j] = f2bf(k1[(j < 8 ? half * 8 + j : 8 + half * 8 + j)]);
            }
        }
        v8f z = vzero8();
        v8f s0 = __builtin_amdgcn_wmma_f32_16x16x32_bf16(false, fq.v, false, fk0.v,
                                                         (short)0, z, false, false);
        v8f s1 = __builtin_amdgcn_wmma_f32_16x16x32_bf16(false, fq.v, false, fk1.v,
                                                         (short)0, z, false, false);
        // online softmax per row (row = r + 8*half, distributed over 16 lanes)
#pragma unroll
        for (int r = 0; r < 8; ++r) {
            float t = fmaxf(s0[r], s1[r]);
            for (int off = 1; off < 16; off <<= 1) t = fmaxf(t, __shfl_xor(t, off, 32));
            float mn = fmaxf(m_run[r], t);
            float p0 = __expf(s0[r] - mn);
            float p1 = __expf(s1[r] - mn);
            float rs = p0 + p1;
            for (int off = 1; off < 16; off <<= 1) rs += __shfl_xor(rs, off, 32);
            float corr = __expf(m_run[r] - mn);
            l_run[r] = l_run[r] * corr + rs;
            m_run[r] = mn;
            o0[r] *= corr;
            o1[r] *= corr;
            P[(r + 8 * half) * 32 + l15]      = f2bf(p0);
            P[(r + 8 * half) * 32 + 16 + l15] = f2bf(p1);
        }
        // bounce P (C layout) -> A fragment layout through per-wave LDS
        FragBF fp;
        {
            const v4u* prow = (const v4u*)(P + l15 * 32);
            fp.u[0] = prow[half];       // K 0-7  / 8-15
            fp.u[1] = prow[2 + half];   // K 16-23 / 24-31
        }

        // ---- wait for the async V tile, build V fragments from LDS ----
        asm volatile("s_wait_asynccnt 0x0" ::: "memory");
        FragBF fv0, fv1;
#pragma unroll
        for (int j = 0; j < 16; ++j) {
            const float* vr = VW + (j < 8 ? half * 8 + j : 8 + half * 8 + j) * 32;
            fv0.s[j] = f2bf(vr[l15]);
            fv1.s[j] = f2bf(vr[16 + l15]);
        }
        o0 = __builtin_amdgcn_wmma_f32_16x16x32_bf16(false, fp.v, false, fv0.v,
                                                     (short)0, o0, false, false);
        o1 = __builtin_amdgcn_wmma_f32_16x16x32_bf16(false, fp.v, false, fv1.v,
                                                     (short)0, o1, false, false);
    }

#pragma unroll
    for (int r = 0; r < 8; ++r) {
        float inv = 1.0f / l_run[r];
        size_t row = (size_t)(b * NN + qtile * 16 + r + 8 * half);
        Oout[row * CC + h * HDD + l15]      = o0[r] * inv;
        Oout[row * CC + h * HDD + 16 + l15] = o1[r] * inv;
    }
}

// ---------------------------------------------------------------------------
// Spectral branch, one block per (b,c) plane:
//   attn_map = real(IF * ((F q F) .* (F k F)) * IF);  out = attn*v + dwconv(v)
// All stages in dynamic LDS (9 * 3136 floats).
// ---------------------------------------------------------------------------
__global__ void k_spectral_plane(const float* __restrict__ Qs,
                                 const float* __restrict__ KVs,
                                 const float* __restrict__ dww,
                                 const float* __restrict__ dwb,
                                 const float* __restrict__ Fr,
                                 const float* __restrict__ Fi,
                                 unsigned short* __restrict__ spre) {
    extern __shared__ float sm[];
    float* RQ  = sm;
    float* RK  = RQ + NN;
    float* RV  = RK + NN;
    float* Ar  = RV + NN;
    float* Ai  = Ar + NN;
    float* QFr = Ai + NN;
    float* QFi = QFr + NN;
    float* PRr = QFi + NN;
    float* PRi = PRr + NN;

    const int p = blockIdx.x;            // plane
    const int b = p / CC;
    const int c = p % CC;

    for (int i = threadIdx.x; i < NN; i += blockDim.x) {
        RQ[i] = Qs[((size_t)(b * NN + i)) * CC + c];
        RK[i] = KVs[((size_t)(b * NN + i)) * 512 + c];
        RV[i] = KVs[((size_t)(b * NN + i)) * 512 + 256 + c];
    }
    __syncthreads();

    // forward y on q
    for (int i = threadIdx.x; i < NN; i += blockDim.x) {
        int ky = i / HH, x = i % HH;
        float sr = 0.f, si = 0.f;
        for (int y = 0; y < HH; ++y) {
            float qv = RQ[y * HH + x];
            sr += Fr[ky * HH + y] * qv;
            si += Fi[ky * HH + y] * qv;
        }
        Ar[i] = sr; Ai[i] = si;
    }
    __syncthreads();
    // forward x on q -> QF
    for (int i = threadIdx.x; i < NN; i += blockDim.x) {
        int ky = i / HH, kx = i % HH;
        float sr = 0.f, si = 0.f;
        for (int x = 0; x < HH; ++x) {
            float tr = Ar[ky * HH + x], ti = Ai[ky * HH + x];
            float fr = Fr[kx * HH + x], fi = Fi[kx * HH + x];
            sr += tr * fr - ti * fi;
            si += tr * fi + ti * fr;
        }
        QFr[i] = sr; QFi[i] = si;
    }
    __syncthreads();
    // forward y on k
    for (int i = threadIdx.x; i < NN; i += blockDim.x) {
        int ky = i / HH, x = i % HH;
        float sr = 0.f, si = 0.f;
        for (int y = 0; y < HH; ++y) {
            float kv = RK[y * HH + x];
            sr += Fr[ky * HH + y] * kv;
            si += Fi[ky * HH + y] * kv;
        }
        Ar[i] = sr; Ai[i] = si;
    }
    __syncthreads();
    // forward x on k, fused product with QF -> PR
    for (int i = threadIdx.x; i < NN; i += blockDim.x) {
        int ky = i / HH, kx = i % HH;
        float sr = 0.f, si = 0.f;
        for (int x = 0; x < HH; ++x) {
            float tr = Ar[ky * HH + x], ti = Ai[ky * HH + x];
            float fr = Fr[kx * HH + x], fi = Fi[kx * HH + x];
            sr += tr * fr - ti * fi;
            si += tr * fi + ti * fr;
        }
        float qr = QFr[i], qi = QFi[i];
        PRr[i] = sr * qr - si * qi;
        PRi[i] = sr * qi + si * qr;
    }
    __syncthreads();
    // inverse y
    for (int i = threadIdx.x; i < NN; i += blockDim.x) {
        int y = i / HH, kx = i % HH;
        float sr = 0.f, si = 0.f;
        for (int ky = 0; ky < HH; ++ky) {
            float pr = PRr[ky * HH + kx], pi = PRi[ky * HH + kx];
            float fr = Fr[y * HH + ky],  fi = Fi[y * HH + ky];
            sr += fr * pr + fi * pi;
            si += fr * pi - fi * pr;
        }
        Ar[i] = sr; Ai[i] = si;
    }
    __syncthreads();
    // inverse x (real part) + v * attn + depthwise 3x3 epilogue
    for (int i = threadIdx.x; i < NN; i += blockDim.x) {
        int y = i / HH, x = i % HH;
        float sr = 0.f;
        for (int kx = 0; kx < HH; ++kx)
            sr += Fr[x * HH + kx] * Ar[y * HH + kx] + Fi[x * HH + kx] * Ai[y * HH + kx];
        float vl = dwb[c];
#pragma unroll
        for (int di = 0; di < 3; ++di)
#pragma unroll
            for (int dj = 0; dj < 3; ++dj) {
                int yy = y - 1 + di, xx = x - 1 + dj;
                if (yy >= 0 && yy < HH && xx >= 0 && xx < HH)
                    vl += dww[c * 9 + di * 3 + dj] * RV[yy * HH + xx];
            }
        spre[((size_t)(b * NN + i)) * CC + c] = f2bf(sr * RV[i] + vl);
    }
}

// ---------------------------------------------------------------------------
// Token branch residual, one block per (b,c) plane:
//   v_res = real(IF * ((F v F) .* Wext) * IF);   tpre = bf16(x_attn + v_res)
// ---------------------------------------------------------------------------
__global__ void k_token_plane(const float* __restrict__ KVt,
                              const float* __restrict__ xat,
                              const float* __restrict__ cw,
                              const float* __restrict__ Fr,
                              const float* __restrict__ Fi,
                              unsigned short* __restrict__ tpre) {
    extern __shared__ float sm[];
    float* RV  = sm;
    float* Ar  = RV + NN;
    float* Ai  = Ar + NN;
    float* VFr = Ai + NN;
    float* VFi = VFr + NN;

    const int p = blockIdx.x;
    const int b = p / CC;
    const int c = p % CC;

    for (int i = threadIdx.x; i < NN; i += blockDim.x)
        RV[i] = KVt[((size_t)(b * NN + i)) * 512 + 256 + c];
    __syncthreads();

    for (int i = threadIdx.x; i < NN; i += blockDim.x) {
        int ky = i / HH, x = i % HH;
        float sr = 0.f, si = 0.f;
        for (int y = 0; y < HH; ++y) {
            float vv = RV[y * HH + x];
            sr += Fr[ky * HH + y] * vv;
            si += Fi[ky * HH + y] * vv;
        }
        Ar[i] = sr; Ai[i] = si;
    }
    __syncthreads();
    // forward x + complex weight multiply (Hermitian extension of half-spectrum)
    for (int i = threadIdx.x; i < NN; i += blockDim.x) {
        int ky = i / HH, kx = i % HH;
        float sr = 0.f, si = 0.f;
        for (int x = 0; x < HH; ++x) {
            float tr = Ar[ky * HH + x], ti = Ai[ky * HH + x];
            float fr = Fr[kx * HH + x], fi = Fi[kx * HH + x];
            sr += tr * fr - ti * fi;
            si += tr * fi + ti * fr;
        }
        float wr, wi;
        if (kx <= 28) {
            const float* wp = cw + (((size_t)c * HH + ky) * 29 + kx) * 2;
            wr = wp[0]; wi = wp[1];
        } else {
            int ky2 = (HH - ky) % HH, kx2 = HH - kx;
            const float* wp = cw + (((size_t)c * HH + ky2) * 29 + kx2) * 2;
            wr = wp[0]; wi = -wp[1];
        }
        VFr[i] = sr * wr - si * wi;
        VFi[i] = sr * wi + si * wr;
    }
    __syncthreads();
    for (int i = threadIdx.x; i < NN; i += blockDim.x) {
        int y = i / HH, kx = i % HH;
        float sr = 0.f, si = 0.f;
        for (int ky = 0; ky < HH; ++ky) {
            float pr = VFr[ky * HH + kx], pi = VFi[ky * HH + kx];
            float fr = Fr[y * HH + ky],  fi = Fi[y * HH + ky];
            sr += fr * pr + fi * pi;
            si += fr * pi - fi * pr;
        }
        Ar[i] = sr; Ai[i] = si;
    }
    __syncthreads();
    for (int i = threadIdx.x; i < NN; i += blockDim.x) {
        int y = i / HH, x = i % HH;
        float sr = 0.f;
        for (int kx = 0; kx < HH; ++kx)
            sr += Fr[x * HH + kx] * Ar[y * HH + kx] + Fi[x * HH + kx] * Ai[y * HH + kx];
        size_t row = (size_t)(b * NN + i);
        tpre[row * CC + c] = f2bf(xat[row * CC + c] + sr);
    }
}

// ---------------------------------------------------------------------------
// Gate + blend. One wave per pixel: gate = sigmoid(w2 . relu(BN(h)) + b2),
// out[b,c,h,w] = gate*os + (1-gate)*ot  (also transposes back to NCHW).
// ---------------------------------------------------------------------------
__global__ void k_gate_fuse(const float* __restrict__ Hh,
                            const float* __restrict__ OS,
                            const float* __restrict__ OT,
                            const float* __restrict__ bng,
                            const float* __restrict__ bnb,
                            const float* __restrict__ w2,
                            const float* __restrict__ b2,
                            float* __restrict__ out) {
    const int lane = threadIdx.x & 31;
    const int wid  = threadIdx.x >> 5;
    const int n    = blockIdx.x * 8 + wid;
    if (n >= MM) return;
    const int b = n / NN, sp = n % NN;
    const float bns = rsqrtf(1.0f + 1e-5f);

    float acc = 0.f;
    for (int c = lane; c < CC; c += 32) {
        float hv = Hh[(size_t)n * CC + c] * (bng[c] * bns) + bnb[c];
        hv = fmaxf(hv, 0.f);
        acc += w2[c] * hv;
    }
    for (int off = 16; off >= 1; off >>= 1) acc += __shfl_xor(acc, off, 32);
    float gate = 1.0f / (1.0f + __expf(-(acc + b2[0])));

    for (int c = lane; c < CC; c += 32) {
        float a = OS[(size_t)n * CC + c];
        float t = OT[(size_t)n * CC + c];
        out[((size_t)(b * CC + c)) * NN + sp] = gate * a + (1.0f - gate) * t;
    }
}

// ---------------------------------------------------------------------------
extern "C" void kernel_launch(void* const* d_in, const int* in_sizes, int n_in,
                              void* d_out, int out_size, void* d_ws, size_t ws_size,
                              hipStream_t stream) {
    const float* x        = (const float*)d_in[0];
    const float* ctx      = (const float*)d_in[1];
    const float* s_q_w    = (const float*)d_in[2];
    const float* s_kv_w   = (const float*)d_in[3];
    const float* s_proj_w = (const float*)d_in[4];
    const float* s_proj_b = (const float*)d_in[5];
    const float* s_dw_w   = (const float*)d_in[6];
    const float* s_dw_b   = (const float*)d_in[7];
    const float* t_q_w    = (const float*)d_in[8];
    const float* t_kv_w   = (const float*)d_in[9];
    const float* t_proj_w = (const float*)d_in[10];
    const float* t_proj_b = (const float*)d_in[11];
    const float* t_cw     = (const float*)d_in[12];
    const float* g_w1     = (const float*)d_in[13];
    const float* g_bn_g   = (const float*)d_in[14];
    const float* g_bn_b   = (const float*)d_in[15];
    const float* g_w2     = (const float*)d_in[16];
    const float* g_b2     = (const float*)d_in[17];
    float* out = (float*)d_out;

    char* wp = (char*)d_ws;
    auto alloc = [&](size_t bytes) -> void* {
        void* r = (void*)wp;
        wp += (bytes + 255) & ~(size_t)255;
        return r;
    };

    unsigned short* xt    = (unsigned short*)alloc((size_t)MM * CC * 2);
    unsigned short* ct    = (unsigned short*)alloc((size_t)MM * CC * 2);
    unsigned short* wsq   = (unsigned short*)alloc(65536 * 2);
    unsigned short* wskv  = (unsigned short*)alloc(131072 * 2);
    unsigned short* wtq   = (unsigned short*)alloc(65536 * 2);
    unsigned short* wtkv  = (unsigned short*)alloc(131072 * 2);
    unsigned short* wsp   = (unsigned short*)alloc(65536 * 2);
    unsigned short* wtp   = (unsigned short*)alloc(65536 * 2);
    unsigned short* wg1   = (unsigned short*)alloc(131072 * 2);
    float* q_s   = (float*)alloc((size_t)MM * 256 * 4);
    float* kv_s  = (float*)alloc((size_t)MM * 512 * 4);
    float* q_t   = (float*)alloc((size_t)MM * 256 * 4);
    float* kv_t  = (float*)alloc((size_t)MM * 512 * 4);
    float* xattn = (float*)alloc((size_t)MM * 256 * 4);
    unsigned short* spre = (unsigned short*)alloc((size_t)MM * CC * 2);
    unsigned short* tpre = (unsigned short*)alloc((size_t)MM * CC * 2);
    float* osb   = (float*)alloc((size_t)MM * 256 * 4);
    float* otb   = (float*)alloc((size_t)MM * 256 * 4);
    unsigned short* osbf = (unsigned short*)alloc((size_t)MM * CC * 2);
    unsigned short* otbf = (unsigned short*)alloc((size_t)MM * CC * 2);
    float* hbuf  = (float*)alloc((size_t)MM * 256 * 4);
    float* Fr    = (float*)alloc((size_t)HH * HH * 4);
    float* Fi    = (float*)alloc((size_t)HH * HH * 4);

    const int T = 256;
    // weight conversions
    k_f32_to_bf16<<<(65536 + T - 1) / T, T, 0, stream>>>(s_q_w, wsq, 65536);
    k_f32_to_bf16<<<(131072 + T - 1) / T, T, 0, stream>>>(s_kv_w, wskv, 131072);
    k_f32_to_bf16<<<(65536 + T - 1) / T, T, 0, stream>>>(t_q_w, wtq, 65536);
    k_f32_to_bf16<<<(131072 + T - 1) / T, T, 0, stream>>>(t_kv_w, wtkv, 131072);
    k_f32_to_bf16<<<(65536 + T - 1) / T, T, 0, stream>>>(s_proj_w, wsp, 65536);
    k_f32_to_bf16<<<(65536 + T - 1) / T, T, 0, stream>>>(t_proj_w, wtp, 65536);
    k_f32_to_bf16<<<(131072 + T - 1) / T, T, 0, stream>>>(g_w1, wg1, 131072);
    // activations -> channels-last bf16
    k_nchw_to_nc<<<(BB * CC * NN + T - 1) / T, T, 0, stream>>>(x, xt);
    k_nchw_to_nc<<<(BB * CC * NN + T - 1) / T, T, 0, stream>>>(ctx, ct);
    k_build_dft<<<(HH * HH + T - 1) / T, T, 0, stream>>>(Fr, Fi);

    dim3 g256(MM / 64, 256 / 32);
    dim3 g512(MM / 64, 512 / 32);
    // projections (WMMA)
    k_gemm_bf16<<<g256, T, 0, stream>>>(xt, 256, wsq, 256, q_s, 256, nullptr, 256, 0);
    k_gemm_bf16<<<g512, T, 0, stream>>>(ct, 256, wskv, 256, kv_s, 512, nullptr, 256, 0);
    k_gemm_bf16<<<g256, T, 0, stream>>>(xt, 256, wtq, 256, q_t, 256, nullptr, 256, 0);
    k_gemm_bf16<<<g512, T, 0, stream>>>(ct, 256, wtkv, 256, kv_t, 512, nullptr, 256, 0);

    // spectral branch (DFT-in-LDS) and fused attention (WMMA + async->LDS)
    k_spectral_plane<<<BB * CC, T, (size_t)9 * NN * 4, stream>>>(
        q_s, kv_s, s_dw_w, s_dw_b, Fr, Fi, spre);
    k_attn<<<392, T, 0, stream>>>(q_t, kv_t, xattn);
    k_token_plane<<<BB * CC, T, (size_t)5 * NN * 4, stream>>>(
        kv_t, xattn, t_cw, Fr, Fi, tpre);

    // output projections (WMMA, with bias)
    k_gemm_bf16<<<g256, T, 0, stream>>>(spre, 256, wsp, 256, osb, 256, s_proj_b, 256, 0);
    k_gemm_bf16<<<g256, T, 0, stream>>>(tpre, 256, wtp, 256, otb, 256, t_proj_b, 256, 0);

    // gate hidden layer: h = os @ w1a^T + ot @ w1b^T  (accumulating WMMA GEMMs)
    k_f32_to_bf16<<<((size_t)MM * CC + T - 1) / T, T, 0, stream>>>(osb, osbf, MM * CC);
    k_f32_to_bf16<<<((size_t)MM * CC + T - 1) / T, T, 0, stream>>>(otb, otbf, MM * CC);
    k_gemm_bf16<<<g256, T, 0, stream>>>(osbf, 256, wg1, 512, hbuf, 256, nullptr, 256, 0);
    k_gemm_bf16<<<g256, T, 0, stream>>>(otbf, 256, wg1 + 256, 512, hbuf, 256, nullptr, 256, 1);

    // gate + blend + NCHW transpose
    k_gate_fuse<<<MM / 8, T, 0, stream>>>(hbuf, osb, otb, g_bn_g, g_bn_b, g_w2, g_b2, out);

    (void)in_sizes; (void)n_in; (void)out_size; (void)ws_size;
}
